// MarkovBlanketAttention_67551245631873
// MI455X (gfx1250) — matrix operand order repaired and checked by reading, not compiled, not run on hardware
//
#include <hip/hip_runtime.h>
#include <hip/hip_bf16.h>

typedef _Float16 h16;
typedef __attribute__((ext_vector_type(16))) _Float16 v16h;
typedef __attribute__((ext_vector_type(8)))  float    v8f;

// Problem constants (from reference)
constexpr int Bc  = 2;
constexpr int Sc  = 1024;
constexpr int DIN = 512;
constexpr int DHc = 1024;
constexpr int Hc  = 16;
constexpr int HDc = 64;
constexpr int Mrows = Bc * Sc;          // 2048 flattened (B,S) rows

// ---------------------------------------------------------------------------
// WMMA 16x16x32 f16 fragment helpers (layouts per CDNA5 ISA §7.12.2, wave32)
// ---------------------------------------------------------------------------

// A fragment: A[m][k], 16x32, src row-major with row stride lda (elements).
// lane L: m = L%16, half = L/16; VGPR i holds K pair k0 = (i/4)*16 + half*8 + (i%4)*2
__device__ __forceinline__ v16h load_A(const h16* __restrict__ src, int lda) {
  const int lane = threadIdx.x & 31;
  const int m = lane & 15, half = lane >> 4;
  v16h a;
#pragma unroll
  for (int i = 0; i < 8; ++i) {
    const int k0 = (i >> 2) * 16 + half * 8 + (i & 3) * 2;
    const h16* p = src + (size_t)m * lda + k0;
    a[2 * i]     = p[0];
    a[2 * i + 1] = p[1];
  }
  return a;
}

// B fragment from B^T storage: bt[n][k] = B[k][n], row stride ldb (elements).
// lane L: n = L%16, half = L/16; VGPR i holds K pair k0 = (i/4)*16 + half*8 + (i%4)*2
__device__ __forceinline__ v16h load_Bt(const h16* __restrict__ bt, int ldb) {
  const int lane = threadIdx.x & 31;
  const int n = lane & 15, half = lane >> 4;
  v16h b;
#pragma unroll
  for (int i = 0; i < 8; ++i) {
    const int k0 = (i >> 2) * 16 + half * 8 + (i & 3) * 2;
    const h16* p = bt + (size_t)n * ldb + k0;
    b[2 * i]     = p[0];
    b[2 * i + 1] = p[1];
  }
  return b;
}

// C/D store: VGPR r, lane L -> M = r + 8*(L/16), N = L%16
__device__ __forceinline__ void store_C_f32(float* __restrict__ dst, int ldc, const v8f& c) {
  const int lane = threadIdx.x & 31;
  const int n = lane & 15, half = lane >> 4;
#pragma unroll
  for (int r = 0; r < 8; ++r)
    dst[(size_t)(r + half * 8) * ldc + n] = c[r];
}

__device__ __forceinline__ v8f wmma_f16(v16h a, v16h b, v8f c) {
  return __builtin_amdgcn_wmma_f32_16x16x32_f16(false, a, false, b, (short)0, c, false, false);
}

// Async copy of 4KB row (1024 f32) global -> LDS via CDNA5 async-to-LDS path.
// Each of 8 instructions moves 32 lanes x 16B = 512B; tracked by ASYNCcnt.
__device__ __forceinline__ void async_row_load(const float* __restrict__ g,
                                               float* lds, int lane) {
#pragma unroll
  for (int c = 0; c < 8; ++c) {
    const float* ga = g + lane * 4 + c * 128;
    unsigned la = (unsigned)(uintptr_t)(lds + lane * 4 + c * 128);  // addr[31:0] = LDS offset
    asm volatile("global_load_async_to_lds_b128 %0, %1, off" :: "v"(la), "v"(ga) : "memory");
  }
}

// ---------------------------------------------------------------------------
// Elementwise conversion / weight prep
// ---------------------------------------------------------------------------
__global__ void k_f32_to_f16(const float* __restrict__ x, h16* __restrict__ y, int n) {
  int i = blockIdx.x * blockDim.x + threadIdx.x;
  if (i < n) y[i] = (h16)x[i];
}

// W [K,N] f32 row-major -> Wt [N,K] f16
__global__ void k_w_transpose(const float* __restrict__ w, h16* __restrict__ wt, int K, int N) {
  int i = blockIdx.x * blockDim.x + threadIdx.x;
  if (i < K * N) {
    int k = i / N, n = i % N;
    wt[(size_t)n * K + k] = (h16)w[i];
  }
}

// ---------------------------------------------------------------------------
// Multi-term projection GEMM with 1x4 N-blocking (one A frag feeds 4 WMMAs):
//   out = sum_t X_t @ W_t + sum_t bias_t
//   X_t: [M,K] f16 row-major,  Wt_t: [N,K] f16 (pre-transposed)
//   OUTMODE 0: f16 head-major [B,H,S,HD]
//   OUTMODE 1: f32 plain [M,N]
//   OUTMODE 2: f16 per-head transposed [B,H,HD,S]   (for V -> coalesced PV B-frags)
// ---------------------------------------------------------------------------
struct GemmArgs {
  const h16*   X[3];
  const h16*   Wt[3];
  const float* bias[3];
  int nterms, M, N, K;
};

template <int OUTMODE>
__global__ void __launch_bounds__(256) k_gemm(GemmArgs ga, void* __restrict__ outp) {
  const int wave = threadIdx.x >> 5;
  const int lane = threadIdx.x & 31;
  const int tiles_n = ga.N >> 6;                       // 64-wide N blocks
  const int tile = blockIdx.x * (blockDim.x >> 5) + wave;
  if (tile >= (ga.M >> 4) * tiles_n) return;           // wave-uniform guard
  const int mt = tile / tiles_n, nt = tile % tiles_n;

  v8f acc[4] = {v8f{}, v8f{}, v8f{}, v8f{}};
  for (int t = 0; t < ga.nterms; ++t) {
    const h16* xrow = ga.X[t]  + (size_t)mt * 16 * ga.K;
    const h16* wrow = ga.Wt[t] + (size_t)nt * 64 * ga.K;
    for (int k0 = 0; k0 < ga.K; k0 += 32) {
      __builtin_prefetch(xrow + k0 + 256, 0, 1);       // global_prefetch_b8
      v16h a = load_A(xrow + k0, ga.K);
#pragma unroll
      for (int j = 0; j < 4; ++j) {
        v16h b = load_Bt(wrow + (size_t)j * 16 * ga.K + k0, ga.K);
        acc[j] = wmma_f16(a, b, acc[j]);
      }
    }
  }

  const int half = lane >> 4;
#pragma unroll
  for (int j = 0; j < 4; ++j) {
    const int n = nt * 64 + j * 16 + (lane & 15);
    float bsum = 0.f;
    for (int t = 0; t < ga.nterms; ++t) bsum += ga.bias[t][n];

    if (OUTMODE == 1) {
      float* o = (float*)outp;
#pragma unroll
      for (int r = 0; r < 8; ++r)
        o[(size_t)(mt * 16 + r + half * 8) * ga.N + n] = acc[j][r] + bsum;
    } else if (OUTMODE == 0) {
      h16* o = (h16*)outp;
      const int h = n / HDc, hd = n % HDc;
#pragma unroll
      for (int r = 0; r < 8; ++r) {
        const int m = mt * 16 + r + half * 8;
        const int bidx = m / Sc, srow = m % Sc;
        o[((((size_t)bidx * Hc + h) * Sc) + srow) * HDc + hd] = (h16)(acc[j][r] + bsum);
      }
    } else {                                           // OUTMODE 2: vT [B,H,HD,S]
      h16* o = (h16*)outp;
      const int h = n / HDc, hd = n % HDc;
#pragma unroll
      for (int r = 0; r < 8; ++r) {
        const int m = mt * 16 + r + half * 8;
        const int bidx = m / Sc, srow = m % Sc;
        o[((((size_t)bidx * Hc + h) * HDc) + hd) * Sc + srow] = (h16)(acc[j][r] + bsum);
      }
    }
  }
}

// ---------------------------------------------------------------------------
// Raw scores with 1x4 key blocking: sc[b,h,i,j] = sum_d q[b,h,i,d]*k[b,h,j,d]
// ---------------------------------------------------------------------------
__global__ void __launch_bounds__(256) k_scores(const h16* __restrict__ q,
                                                const h16* __restrict__ k,
                                                float* __restrict__ sc) {
  const int wave = threadIdx.x >> 5;
  const int tiles_per_head = (Sc / 16) * (Sc / 64);    // 1024
  const int tile = blockIdx.x * (blockDim.x >> 5) + wave;
  if (tile >= Bc * Hc * tiles_per_head) return;        // wave-uniform
  const int bh = tile / tiles_per_head;
  const int t  = tile % tiles_per_head;
  const int it = t / (Sc / 64), jt = t % (Sc / 64);

  const h16* qh = q + (size_t)bh * Sc * HDc + (size_t)it * 16 * HDc;
  const h16* kh = k + (size_t)bh * Sc * HDc + (size_t)jt * 64 * HDc;

  v8f acc[4] = {v8f{}, v8f{}, v8f{}, v8f{}};
#pragma unroll
  for (int k0 = 0; k0 < HDc; k0 += 32) {
    v16h a = load_A(qh + k0, HDc);
#pragma unroll
    for (int j = 0; j < 4; ++j) {
      v16h b = load_Bt(kh + (size_t)j * 16 * HDc + k0, HDc);  // k rows act as B^T
      acc[j] = wmma_f16(a, b, acc[j]);
    }
  }
  float* dst = sc + (size_t)bh * Sc * Sc + (size_t)it * 16 * Sc + (size_t)jt * 64;
#pragma unroll
  for (int j = 0; j < 4; ++j) store_C_f32(dst + j * 16, Sc, acc[j]);
}

// ---------------------------------------------------------------------------
// Softmax over (scores+bias) rows, then P@V via WMMA.
// Score/bias rows are double-buffered into LDS with global_load_async_to_lds_b128
// (ASYNCcnt), overlapping the next row's fetch with this row's exp math.
// V is pre-transposed per head ([B,H,HD,S]) so PV B-fragments are contiguous.
// 4 waves per block; each wave owns one 16-row query tile of one (b,h).
// ---------------------------------------------------------------------------
__global__ void __launch_bounds__(128) k_softmax_pv(const float* __restrict__ sc,
                                                    const float* __restrict__ bias,
                                                    const h16* __restrict__ vT,
                                                    h16* __restrict__ o16) {
  __shared__ h16 probs[4][16][Sc];                      // 128 KB
  __shared__ __align__(16) float stage[4][2][2][Sc];    // 64 KB (dbl-buf x {score,bias})
  const int wave = threadIdx.x >> 5;
  const int lane = threadIdx.x & 31;
  const int qtile = blockIdx.x * 4 + wave;              // over B*H*(S/16) = 2048
  if (qtile >= Bc * Hc * (Sc / 16)) return;
  const int qt = qtile % (Sc / 16);
  const int bh = qtile / (Sc / 16);
  const int bidx = bh / Hc, h = bh % Hc;

  const float* sbase = sc   + (size_t)bh * Sc * Sc + (size_t)qt * 16 * Sc;
  const float* bbase = bias + (size_t)bh * Sc * Sc + (size_t)qt * 16 * Sc;

  // Prime row 0.
  async_row_load(sbase, stage[wave][0][0], lane);
  async_row_load(bbase, stage[wave][0][1], lane);

  for (int r = 0; r < 16; ++r) {
    if (r < 15) {                                       // prefetch next row (16 async ops)
      async_row_load(sbase + (size_t)(r + 1) * Sc, stage[wave][(r + 1) & 1][0], lane);
      async_row_load(bbase + (size_t)(r + 1) * Sc, stage[wave][(r + 1) & 1][1], lane);
      asm volatile("s_wait_asynccnt 0x10" ::: "memory"); // current row done, next in flight
    } else {
      asm volatile("s_wait_asynccnt 0x0" ::: "memory");
    }
    const float* stS = stage[wave][r & 1][0];
    const float* stB = stage[wave][r & 1][1];

    float xs[Sc / 32];
    float mx = -3.0e38f;
#pragma unroll
    for (int c = 0; c < Sc / 32; ++c) {
      float xv = stS[lane + c * 32] + stB[lane + c * 32];
      xs[c] = xv;
      mx = fmaxf(mx, xv);
    }
#pragma unroll
    for (int off = 16; off > 0; off >>= 1) mx = fmaxf(mx, __shfl_xor(mx, off, 32));
    float sum = 0.f;
#pragma unroll
    for (int c = 0; c < Sc / 32; ++c) { xs[c] = __expf(xs[c] - mx); sum += xs[c]; }
#pragma unroll
    for (int off = 16; off > 0; off >>= 1) sum += __shfl_xor(sum, off, 32);
    const float inv = 1.0f / sum;
#pragma unroll
    for (int c = 0; c < Sc / 32; ++c)
      probs[wave][r][lane + c * 32] = (h16)(xs[c] * inv);
  }

  // P @ V : K = S = 1024 in 32-wide steps; HD=64 -> 4 N-tiles; vT gives coalesced B.
  const h16* vh = vT + (size_t)bh * HDc * Sc;
  v8f acc[4] = {v8f{}, v8f{}, v8f{}, v8f{}};
  for (int kt = 0; kt < Sc / 32; ++kt) {
    v16h a = load_A(&probs[wave][0][0] + kt * 32, Sc);  // ds_load fragments
#pragma unroll
    for (int nt = 0; nt < 4; ++nt) {
      v16h bf = load_Bt(vh + (size_t)(nt * 16) * Sc + kt * 32, Sc);
      acc[nt] = wmma_f16(a, bf, acc[nt]);
    }
  }

  const int n = lane & 15, half = lane >> 4;
#pragma unroll
  for (int nt = 0; nt < 4; ++nt)
#pragma unroll
    for (int r = 0; r < 8; ++r) {
      const int i = qt * 16 + r + half * 8;
      o16[((size_t)bidx * Sc + i) * DHc + h * HDc + nt * 16 + n] = (h16)acc[nt][r];
    }
}

// ---------------------------------------------------------------------------
// Residual chain: s_out = s + Ps ; b_out = s_out + b + Pb ; z_out = b_out + z + Pz
// (Ps/Pb/Pz already include the co bias from the Wo GEMM.)
// ---------------------------------------------------------------------------
__global__ void k_residual(const float* __restrict__ s, const float* __restrict__ b,
                           const float* __restrict__ z,
                           const float* __restrict__ Ps, const float* __restrict__ Pb,
                           const float* __restrict__ Pz,
                           float* __restrict__ s_out, float* __restrict__ b_out,
                           float* __restrict__ z_out, int n) {
  int i = blockIdx.x * blockDim.x + threadIdx.x;
  if (i >= n) return;
  float so = s[i] + Ps[i];
  float bo = so + b[i] + Pb[i];
  float zo = bo + z[i] + Pz[i];
  s_out[i] = so; b_out[i] = bo; z_out[i] = zo;
}

// ---------------------------------------------------------------------------
// Host side
// ---------------------------------------------------------------------------
static void launch_gemm(int outmode,
                        const h16* x0, const h16* w0, const float* b0,
                        const h16* x1, const h16* w1, const float* b1,
                        const h16* x2, const h16* w2, const float* b2,
                        int nterms, int K, void* out, hipStream_t stream) {
  GemmArgs ga{};
  ga.X[0] = x0; ga.X[1] = x1; ga.X[2] = x2;
  ga.Wt[0] = w0; ga.Wt[1] = w1; ga.Wt[2] = w2;
  ga.bias[0] = b0; ga.bias[1] = b1; ga.bias[2] = b2;
  ga.nterms = nterms; ga.M = Mrows; ga.N = DHc; ga.K = K;
  const int tiles = (Mrows / 16) * (DHc / 64);   // 2048 wave-tiles
  const int grid = tiles / 8;
  if (outmode == 0)      k_gemm<0><<<grid, 256, 0, stream>>>(ga, out);
  else if (outmode == 1) k_gemm<1><<<grid, 256, 0, stream>>>(ga, out);
  else                   k_gemm<2><<<grid, 256, 0, stream>>>(ga, out);
}

extern "C" void kernel_launch(void* const* d_in, const int* in_sizes, int n_in,
                              void* d_out, int out_size, void* d_ws, size_t ws_size,
                              hipStream_t stream) {
  // --- inputs (setup_inputs dict order) ---
  const float* y   = (const float*)d_in[0];
  const float* s   = (const float*)d_in[1];
  const float* b   = (const float*)d_in[2];
  const float* z   = (const float*)d_in[3];
  const float* bias_s = (const float*)d_in[4];
  const float* bias_b = (const float*)d_in[5];
  const float* bias_z = (const float*)d_in[6];
  const float* Wqs = (const float*)d_in[7];   const float* cqs = (const float*)d_in[8];
  const float* Wqb = (const float*)d_in[9];   const float* cqb = (const float*)d_in[10];
  const float* Wqz = (const float*)d_in[11];  const float* cqz = (const float*)d_in[12];
  const float* Wkss = (const float*)d_in[13]; const float* ckss = (const float*)d_in[14];
  const float* Wksb = (const float*)d_in[15]; const float* cksb = (const float*)d_in[16];
  const float* Wkzb = (const float*)d_in[17]; const float* ckzb = (const float*)d_in[18];
  const float* Wkbb = (const float*)d_in[19]; const float* ckbb = (const float*)d_in[20];
  const float* Wkzz = (const float*)d_in[21]; const float* ckzz = (const float*)d_in[22];
  const float* Wvs = (const float*)d_in[23];  const float* cvs = (const float*)d_in[24];
  const float* Wvb = (const float*)d_in[25];  const float* cvb = (const float*)d_in[26];
  const float* Wvz = (const float*)d_in[27];  const float* cvz = (const float*)d_in[28];
  const float* Wo  = (const float*)d_in[29];  const float* co  = (const float*)d_in[30];

  // --- output layout (flat, return order) ---
  const size_t nBSD = (size_t)Bc * Sc * DHc;          // 2,097,152
  const size_t nSC  = (size_t)Bc * Hc * Sc * Sc;      // 33,554,432
  float* out_s  = (float*)d_out;
  float* out_b  = out_s + nBSD;
  float* out_z  = out_b + nBSD;
  float* out_ss = out_z + nBSD;
  float* out_sb = out_ss + nSC;
  float* out_sz = out_sb + nSC;

  // --- workspace carve-up ---
  char* p = (char*)d_ws;
  auto alloc = [&](size_t bytes) { char* r = p; p += (bytes + 255) & ~(size_t)255; return r; };
  h16* y16 = (h16*)alloc((size_t)Bc * Sc * DIN * 2);
  h16* s16 = (h16*)alloc(nBSD * 2);
  h16* b16 = (h16*)alloc(nBSD * 2);
  h16* z16 = (h16*)alloc(nBSD * 2);
  h16* WqsT = (h16*)alloc((size_t)DIN * DHc * 2);
  h16* WqbT = (h16*)alloc((size_t)DIN * DHc * 2);
  h16* WqzT = (h16*)alloc((size_t)DIN * DHc * 2);
  h16* WkssT = (h16*)alloc((size_t)DHc * DHc * 2);
  h16* WksbT = (h16*)alloc((size_t)DHc * DHc * 2);
  h16* WkzbT = (h16*)alloc((size_t)DHc * DHc * 2);
  h16* WkbbT = (h16*)alloc((size_t)DHc * DHc * 2);
  h16* WkzzT = (h16*)alloc((size_t)DHc * DHc * 2);
  h16* WvsT = (h16*)alloc((size_t)DHc * DHc * 2);
  h16* WvbT = (h16*)alloc((size_t)DHc * DHc * 2);
  h16* WvzT = (h16*)alloc((size_t)DHc * DHc * 2);
  h16* WoT  = (h16*)alloc((size_t)DHc * DHc * 2);
  h16* q_s = (h16*)alloc(nBSD * 2);  h16* q_b = (h16*)alloc(nBSD * 2);  h16* q_z = (h16*)alloc(nBSD * 2);
  h16* k_s = (h16*)alloc(nBSD * 2);  h16* k_b = (h16*)alloc(nBSD * 2);  h16* k_z = (h16*)alloc(nBSD * 2);
  h16* v_s = (h16*)alloc(nBSD * 2);  h16* v_b = (h16*)alloc(nBSD * 2);  h16* v_z = (h16*)alloc(nBSD * 2);
  h16* o_s = (h16*)alloc(nBSD * 2);  h16* o_b = (h16*)alloc(nBSD * 2);  h16* o_z = (h16*)alloc(nBSD * 2);
  float* P_s = (float*)alloc(nBSD * 4);
  float* P_b = (float*)alloc(nBSD * 4);
  float* P_z = (float*)alloc(nBSD * 4);

  // 1) activation conversion
  {
    int nY = Bc * Sc * DIN, nX = (int)nBSD;
    k_f32_to_f16<<<(nY + 255) / 256, 256, 0, stream>>>(y, y16, nY);
    k_f32_to_f16<<<(nX + 255) / 256, 256, 0, stream>>>(s, s16, nX);
    k_f32_to_f16<<<(nX + 255) / 256, 256, 0, stream>>>(b, b16, nX);
    k_f32_to_f16<<<(nX + 255) / 256, 256, 0, stream>>>(z, z16, nX);
  }
  // 2) weight transpose + f16 conversion
  {
    int nq = DIN * DHc, nk = DHc * DHc;
    int gq = (nq + 255) / 256, gk = (nk + 255) / 256;
    k_w_transpose<<<gq, 256, 0, stream>>>(Wqs, WqsT, DIN, DHc);
    k_w_transpose<<<gq, 256, 0, stream>>>(Wqb, WqbT, DIN, DHc);
    k_w_transpose<<<gq, 256, 0, stream>>>(Wqz, WqzT, DIN, DHc);
    k_w_transpose<<<gk, 256, 0, stream>>>(Wkss, WkssT, DHc, DHc);
    k_w_transpose<<<gk, 256, 0, stream>>>(Wksb, WksbT, DHc, DHc);
    k_w_transpose<<<gk, 256, 0, stream>>>(Wkzb, WkzbT, DHc, DHc);
    k_w_transpose<<<gk, 256, 0, stream>>>(Wkbb, WkbbT, DHc, DHc);
    k_w_transpose<<<gk, 256, 0, stream>>>(Wkzz, WkzzT, DHc, DHc);
    k_w_transpose<<<gk, 256, 0, stream>>>(Wvs, WvsT, DHc, DHc);
    k_w_transpose<<<gk, 256, 0, stream>>>(Wvb, WvbT, DHc, DHc);
    k_w_transpose<<<gk, 256, 0, stream>>>(Wvz, WvzT, DHc, DHc);
    k_w_transpose<<<gk, 256, 0, stream>>>(Wo, WoT, DHc, DHc);
  }
  // 3) projections (WMMA): q/k head-major [B,H,S,HD]; v per-head transposed [B,H,HD,S]
  launch_gemm(0, y16, WqsT, cqs, 0, 0, 0, 0, 0, 0, 1, DIN, q_s, stream);
  launch_gemm(0, y16, WqbT, cqb, 0, 0, 0, 0, 0, 0, 1, DIN, q_b, stream);
  launch_gemm(0, y16, WqzT, cqz, 0, 0, 0, 0, 0, 0, 1, DIN, q_z, stream);
  launch_gemm(0, s16, WkssT, ckss, b16, WksbT, cksb, 0, 0, 0, 2, DHc, k_s, stream);
  launch_gemm(0, b16, WkbbT, ckbb, s16, WksbT, cksb, z16, WkzbT, ckzb, 3, DHc, k_b, stream);
  launch_gemm(0, z16, WkzzT, ckzz, b16, WkzbT, ckzb, 0, 0, 0, 2, DHc, k_z, stream);
  launch_gemm(2, s16, WvsT, cvs, 0, 0, 0, 0, 0, 0, 1, DHc, v_s, stream);
  launch_gemm(2, b16, WvbT, cvb, 0, 0, 0, 0, 0, 0, 1, DHc, v_b, stream);
  launch_gemm(2, z16, WvzT, cvz, 0, 0, 0, 0, 0, 0, 1, DHc, v_z, stream);

  // 4) raw scores -> d_out (WMMA, 1x4 blocked)
  {
    const int tiles = Bc * Hc * (Sc / 16) * (Sc / 64);  // 32768 wave-tiles
    const int grid = tiles / 8;
    k_scores<<<grid, 256, 0, stream>>>(q_s, k_s, out_ss);
    k_scores<<<grid, 256, 0, stream>>>(q_b, k_b, out_sb);
    k_scores<<<grid, 256, 0, stream>>>(q_z, k_z, out_sz);
  }
  // 5) softmax + PV (async LDS staging + WMMA)
  {
    const int grid = (Bc * Hc * (Sc / 16)) / 4;         // 512
    k_softmax_pv<<<grid, 128, 0, stream>>>(out_ss, bias_s, v_s, o_s);
    k_softmax_pv<<<grid, 128, 0, stream>>>(out_sb, bias_b, v_b, o_b);
    k_softmax_pv<<<grid, 128, 0, stream>>>(out_sz, bias_z, v_z, o_z);
  }
  // 6) output projection P = o @ Wo + co (WMMA, f32 out)
  launch_gemm(1, o_s, WoT, co, 0, 0, 0, 0, 0, 0, 1, DHc, P_s, stream);
  launch_gemm(1, o_b, WoT, co, 0, 0, 0, 0, 0, 0, 1, DHc, P_b, stream);
  launch_gemm(1, o_z, WoT, co, 0, 0, 0, 0, 0, 0, 1, DHc, P_z, stream);

  // 7) residual chain
  {
    int n = (int)nBSD;
    k_residual<<<(n + 255) / 256, 256, 0, stream>>>(s, b, z, P_s, P_b, P_z,
                                                    out_s, out_b, out_z, n);
  }
}
